// MultiScaleMultiHeadedAttention_62783831933086
// MI455X (gfx1250) — compile-verified, matrix operand order checked by reading, not run on hardware
//
#include <hip/hip_runtime.h>

// ---------------------------------------------------------------------------
// MultiScaleMultiHeadedAttention for MI455X (gfx1250, wave32, WMMA bf16)
// B=8, N=512, E=256, H=8, DK=32. All GEMMs via v_wmma_f32_16x16x32_bf16.
// ---------------------------------------------------------------------------

typedef __attribute__((ext_vector_type(16))) __bf16 v16bf;
typedef __attribute__((ext_vector_type(8)))  __bf16 v8bf;
typedef __attribute__((ext_vector_type(8)))  float  v8f;

#define B_   8
#define N_   512
#define E_   256
#define H_   8
#define DK_  32
#define MTOT 4096          // B*N
#define NEGV (-1e10f)

// ---- WMMA helpers ----------------------------------------------------------

static __device__ __forceinline__ v8f wmma_bf16(v16bf a, v16bf b, v8f c) {
  // (neg_a, A, neg_b, B, c_mod, C, reuse_a, reuse_b)
  return __builtin_amdgcn_wmma_f32_16x16x32_bf16(false, a, false, b, (short)0, c,
                                                 false, false);
}

// Load a 16-element bf16 fragment for a lane whose K-elements are contiguous
// in memory in two 8-element halves: [g*8 .. g*8+7] and [16+g*8 .. 16+g*8+7].
// Matches the ISA 16-bit A-matrix (and transposed-B) lane layout.
static __device__ __forceinline__ v16bf frag_ld(const __bf16* rowp, int g) {
  const v8bf lo = *reinterpret_cast<const v8bf*>(rowp + g * 8);
  const v8bf hi = *reinterpret_cast<const v8bf*>(rowp + g * 8 + 16);
  return __builtin_shufflevector(lo, hi, 0, 1, 2, 3, 4, 5, 6, 7,
                                 8, 9, 10, 11, 12, 13, 14, 15);
}

// Pre-packed weight fragment: 16 contiguous bf16 per (ktile, ntile, lane).
static __device__ __forceinline__ v16bf frag_ld_packed(const __bf16* packed,
                                                       int kt, int nt, int NT,
                                                       int lane) {
  const __bf16* p = packed + (((size_t)(kt * NT + nt) * 32 + lane) << 4);
  const v8bf lo = *reinterpret_cast<const v8bf*>(p);
  const v8bf hi = *reinterpret_cast<const v8bf*>(p + 8);
  return __builtin_shufflevector(lo, hi, 0, 1, 2, 3, 4, 5, 6, 7,
                                 8, 9, 10, 11, 12, 13, 14, 15);
}

// XOR-lane reduction helper (ds_swizzle_b32, group-of-32 xor mode).
template <int M>
static __device__ __forceinline__ float swz_xor(float v) {
  return __int_as_float(
      __builtin_amdgcn_ds_swizzle(__float_as_int(v), (M << 10) | 0x1f));
}

// ---- elementwise conversion ------------------------------------------------

__global__ void f32_to_bf16_kernel(const float* __restrict__ in,
                                   __bf16* __restrict__ out, int n) {
  int i = blockIdx.x * blockDim.x + threadIdx.x;
  if (i < n) out[i] = (__bf16)in[i];
}

// ---- weight packing: W (K x N, f32 row-major) -> B-fragment-native bf16 ----

__global__ void pack_w_kernel(const float* __restrict__ W,
                              __bf16* __restrict__ P, int K, int Nn) {
  int t = blockIdx.x * blockDim.x + threadIdx.x;
  int NT = Nn >> 4;
  int KT = K >> 5;
  if (t >= KT * NT * 32) return;
  int lane = t & 31;
  int nt = (t >> 5) % NT;
  int kt = (t >> 5) / NT;
  int g = lane >> 4;
  int n = nt * 16 + (lane & 15);
  __bf16* dst = P + ((size_t)t << 4);
#pragma unroll
  for (int e = 0; e < 16; ++e) {
    int k = kt * 32 + ((e < 8) ? 0 : 16) + g * 8 + (e & 7);
    dst[e] = (__bf16)W[(size_t)k * Nn + n];
  }
}

// ---- projection GEMM: (4096 x 256) @ (256 x 256) + bias, head-scattered ----

__global__ void __launch_bounds__(256)
gemm_heads_kernel(const __bf16* __restrict__ A, const __bf16* __restrict__ Bp,
                  const float* __restrict__ bias, __bf16* __restrict__ out,
                  int vmode) {
  const int lane = threadIdx.x & 31;
  const int wave = threadIdx.x >> 5;
  const int g = lane >> 4, l15 = lane & 15;
  const int mtile = blockIdx.y * 8 + wave;
  const int nt = blockIdx.x;  // NT = 16
  const int m0 = mtile * 16;
  v8f acc = {};
#pragma unroll
  for (int kt = 0; kt < 8; ++kt) {
    v16bf a = frag_ld(A + (size_t)(m0 + l15) * E_ + kt * 32, g);
    v16bf b = frag_ld_packed(Bp, kt, nt, 16, lane);
    acc = wmma_bf16(a, b, acc);
  }
  const int n = nt * 16 + l15;
  const int h = n >> 5, d = n & 31;
  const float bn = bias[n];
#pragma unroll
  for (int r = 0; r < 8; ++r) {
    int m = m0 + r + g * 8;      // flattened b*N + i
    int bb = m >> 9, i = m & 511;
    float v = acc[r] + bn;
    if (vmode)  // V stored transposed: (B,H,DK,N)
      out[(((size_t)bb * H_ + h) * DK_ + d) * N_ + i] = (__bf16)v;
    else        // Q/K: (B,H,N,DK)
      out[(((size_t)bb * H_ + h) * N_ + i) * DK_ + d] = (__bf16)v;
  }
}

// ---- generic flat GEMM: out = relu?(A @ Bp + bias), bf16 or f32 out -------

template <bool RELU, bool F32OUT>
__global__ void __launch_bounds__(256)
gemm_flat_kernel(const __bf16* __restrict__ A, int lda, int K,
                 const __bf16* __restrict__ Bp, int NT,
                 const float* __restrict__ bias, void* __restrict__ outp,
                 int ldo, int ncol_off) {
  const int lane = threadIdx.x & 31;
  const int wave = threadIdx.x >> 5;
  const int g = lane >> 4, l15 = lane & 15;
  const int mtile = blockIdx.y * 8 + wave;
  const int nt = blockIdx.x;
  const int m0 = mtile * 16;
  v8f acc = {};
  for (int kt = 0; kt < (K >> 5); ++kt) {
    v16bf a = frag_ld(A + (size_t)(m0 + l15) * lda + kt * 32, g);
    v16bf b = frag_ld_packed(Bp, kt, nt, NT, lane);
    acc = wmma_bf16(a, b, acc);
  }
  const int n = nt * 16 + l15;
  const float bn = bias[n];
#pragma unroll
  for (int r = 0; r < 8; ++r) {
    int m = m0 + r + g * 8;
    float v = acc[r] + bn;
    if (RELU) v = fmaxf(v, 0.f);
    if (F32OUT)
      ((float*)outp)[(size_t)m * ldo + ncol_off + n] = v;
    else
      ((__bf16*)outp)[(size_t)m * ldo + ncol_off + n] = (__bf16)v;
  }
}

// ---- fused attention: scores + dist_conv + mask + 3x softmax + PV ---------
// Grid: (N/16 query tiles, B*H). Block: 128 threads (4 waves).

__global__ void __launch_bounds__(128)
attn_kernel(const __bf16* __restrict__ qh, const __bf16* __restrict__ kh,
            const __bf16* __restrict__ vT, const float* __restrict__ dist,
            const int* __restrict__ mask, const float* __restrict__ cw1,
            const float* __restrict__ cb1, const float* __restrict__ cw2,
            const float* __restrict__ cb2, __bf16* __restrict__ oB) {
  __shared__ __align__(16) float  sS[16][512];   // masked scaled scores
  __shared__ __align__(16) __bf16 sP[16][512];   // unnormalized exp(p)
  __shared__ float sSum[16];                     // per-row softmax sums

  const int lane = threadIdx.x & 31;
  const int wave = threadIdx.x >> 5;
  const int g = lane >> 4, l15 = lane & 15;
  const int bh = blockIdx.y, b = bh >> 3, h = bh & 7;
  const int q0 = blockIdx.x * 16;

  float w1[8], b1[8], w2[8];
#pragma unroll
  for (int t = 0; t < 8; ++t) {
    w1[t] = cw1[t];
    b1[t] = cb1[t];
    w2[t] = cw2[h * 8 + t];
  }
  const float b2 = cb2[h];
  const float scale = 0.17677669529663687f;  // 1/sqrt(32)

  // -------- scores: 16 x 512 strip, one 16x16 WMMA tile per col-tile -------
  const __bf16* qrow = qh + ((size_t)bh * N_ + q0 + l15) * DK_;
  const v16bf afrag = frag_ld(qrow, g);  // loop-invariant A (K = DK = 32)
  for (int ct = wave; ct < 32; ct += 4) {
    const __bf16* krow = kh + ((size_t)bh * N_ + ct * 16 + l15) * DK_;
    v16bf bfrag = frag_ld(krow, g);
    v8f c = {};
    c = wmma_bf16(afrag, bfrag, c);
    const int j = ct * 16 + l15;
#pragma unroll
    for (int r = 0; r < 8; ++r) {
      int il = r + g * 8;
      int i = q0 + il;
      size_t idx = ((size_t)b * N_ + i) * N_ + j;
      float dv = dist[idx];
      float dc = b2;
#pragma unroll
      for (int t = 0; t < 8; ++t)
        dc += w2[t] * fmaxf(dv * w1[t] + b1[t], 0.f);
      float s = (mask[idx] == 0) ? NEGV : (c[r] * scale) * dc;
      sS[il][j] = s;
    }
  }
  __syncthreads();

  const float bars[3] = {0.3f, 0.7f, 1e10f};
  for (int bar = 0; bar < 3; ++bar) {
    const float bv = bars[bar];
    const int r = threadIdx.x >> 3;     // row 0..15 (8 lanes per row)
    const int cpart = threadIdx.x & 7;
    const int i = q0 + r;

    // pass 1: row max over distance-gated scores
    float mx = -3.4e38f;
    for (int jj = 0; jj < 64; ++jj) {
      int j = cpart * 64 + jj;
      size_t idx = ((size_t)b * N_ + i) * N_ + j;
      bool gate = (dist[idx] < bv) || (i == 0) || (j == 0);
      mx = fmaxf(mx, gate ? sS[r][j] : NEGV);
    }
    mx = fmaxf(mx, swz_xor<1>(mx));
    mx = fmaxf(mx, swz_xor<2>(mx));
    mx = fmaxf(mx, swz_xor<4>(mx));

    // pass 2: exp + sum; store unnormalized exp to sP (bf16)
    float sum = 0.f;
    for (int jj = 0; jj < 64; ++jj) {
      int j = cpart * 64 + jj;
      size_t idx = ((size_t)b * N_ + i) * N_ + j;
      bool gate = (dist[idx] < bv) || (i == 0) || (j == 0);
      float s = gate ? sS[r][j] : NEGV;
      float e = __expf(s - mx);
      sum += e;
      sP[r][j] = (__bf16)e;
    }
    sum += swz_xor<1>(sum);
    sum += swz_xor<2>(sum);
    sum += swz_xor<4>(sum);
    if (cpart == 0) sSum[r] = sum;
    __syncthreads();

    // P (16x512, LDS) @ V (512x32): waves 0/1 each own 16 output columns
    if (wave < 2) {
      v8f acc = {};
      const int d = wave * 16 + l15;
      const __bf16* vrow = vT + ((size_t)bh * DK_ + d) * N_;
#pragma unroll
      for (int kt = 0; kt < 16; ++kt) {
        v16bf ap = frag_ld(&sP[l15][kt * 32], g);      // ds_read from LDS
        v16bf bp = frag_ld(vrow + kt * 32, g);         // contiguous global
        acc = wmma_bf16(ap, bp, acc);
      }
#pragma unroll
      for (int rr = 0; rr < 8; ++rr) {
        int il = rr + g * 8;
        float v = acc[rr] / sSum[il];                  // fold in normalization
        int i2 = q0 + il;
        oB[((size_t)bar * MTOT + (size_t)b * N_ + i2) * E_ + h * DK_ + d] =
            (__bf16)v;
      }
    }
    __syncthreads();  // protect sP/sSum reuse across bars
  }
}

// ---------------------------------------------------------------------------

extern "C" void kernel_launch(void* const* d_in, const int* in_sizes, int n_in,
                              void* d_out, int out_size, void* d_ws,
                              size_t ws_size, hipStream_t stream) {
  (void)in_sizes; (void)n_in; (void)out_size; (void)ws_size;
  const float* query = (const float*)d_in[0];
  const float* key   = (const float*)d_in[1];
  const float* value = (const float*)d_in[2];
  const float* dist  = (const float*)d_in[3];
  const int*   mask  = (const int*)d_in[4];
  const float* Wq = (const float*)d_in[5];  const float* bq = (const float*)d_in[6];
  const float* Wk = (const float*)d_in[7];  const float* bk = (const float*)d_in[8];
  const float* Wv = (const float*)d_in[9];  const float* bv = (const float*)d_in[10];
  const float* Wo = (const float*)d_in[11]; const float* bo = (const float*)d_in[12];
  const float* cw1 = (const float*)d_in[13]; const float* cb1 = (const float*)d_in[14];
  const float* cw2 = (const float*)d_in[15]; const float* cb2 = (const float*)d_in[16];
  const float* Ws1 = (const float*)d_in[17]; const float* bs1 = (const float*)d_in[18];
  const float* Ws2 = (const float*)d_in[19]; const float* bs2 = (const float*)d_in[20];
  float* out = (float*)d_out;

  // workspace carve-up (all bf16 intermediates; ~28 MB total, L2-resident)
  char* ws = (char*)d_ws;
  size_t off = 0;
  auto carve = [&](size_t bytes) -> char* {
    char* p = ws + off;
    off = (off + bytes + 255) & ~(size_t)255;
    return p;
  };
  const size_t SZ_ACT  = (size_t)MTOT * E_ * sizeof(__bf16);      // 2 MB
  const size_t SZ_W    = (size_t)E_ * E_ * sizeof(__bf16);        // 128 KB
  const size_t SZ_W1   = (size_t)3 * E_ * E_ * sizeof(__bf16);    // 384 KB
  __bf16* qA   = (__bf16*)carve(SZ_ACT);
  __bf16* kA   = (__bf16*)carve(SZ_ACT);
  __bf16* vA   = (__bf16*)carve(SZ_ACT);
  __bf16* WqP  = (__bf16*)carve(SZ_W);
  __bf16* WkP  = (__bf16*)carve(SZ_W);
  __bf16* WvP  = (__bf16*)carve(SZ_W);
  __bf16* WoP  = (__bf16*)carve(SZ_W);
  __bf16* Ws2P = (__bf16*)carve(SZ_W);
  __bf16* Ws1P = (__bf16*)carve(SZ_W1);
  __bf16* qh   = (__bf16*)carve(SZ_ACT);   // (B,H,N,DK)
  __bf16* kh   = (__bf16*)carve(SZ_ACT);   // (B,H,N,DK)
  __bf16* vT   = (__bf16*)carve(SZ_ACT);   // (B,H,DK,N)
  __bf16* oB   = (__bf16*)carve(3 * SZ_ACT);              // 3 x (B*N, E)
  __bf16* xB   = (__bf16*)carve((size_t)MTOT * 3 * E_ * 2);  // (B*N, 3E)
  __bf16* h1   = (__bf16*)carve(SZ_ACT);

  const int nact = MTOT * E_;
  // 1) activations -> bf16
  f32_to_bf16_kernel<<<(nact + 255) / 256, 256, 0, stream>>>(query, qA, nact);
  f32_to_bf16_kernel<<<(nact + 255) / 256, 256, 0, stream>>>(key, kA, nact);
  f32_to_bf16_kernel<<<(nact + 255) / 256, 256, 0, stream>>>(value, vA, nact);

  // 2) pack weights into B-fragment layout
  auto packn = [&](const float* W, __bf16* P, int K, int Nn) {
    int cnt = (K >> 5) * (Nn >> 4) * 32;
    pack_w_kernel<<<(cnt + 255) / 256, 256, 0, stream>>>(W, P, K, Nn);
  };
  packn(Wq, WqP, E_, E_);
  packn(Wk, WkP, E_, E_);
  packn(Wv, WvP, E_, E_);
  packn(Wo, WoP, E_, E_);
  packn(Ws2, Ws2P, E_, E_);
  packn(Ws1, Ws1P, 3 * E_, E_);

  // 3) Q/K/V projections (4096x256 @ 256x256) with head scatter
  dim3 gp(16, 32), bp(256);
  gemm_heads_kernel<<<gp, bp, 0, stream>>>(qA, WqP, bq, qh, 0);
  gemm_heads_kernel<<<gp, bp, 0, stream>>>(kA, WkP, bk, kh, 0);
  gemm_heads_kernel<<<gp, bp, 0, stream>>>(vA, WvP, bv, vT, 1);

  // 4) fused attention (scores + dist_conv + mask + 3 bars softmax + PV)
  attn_kernel<<<dim3(N_ / 16, B_ * H_), 128, 0, stream>>>(
      qh, kh, vT, dist, mask, cw1, cb1, cw2, cb2, oB);

  // 5) per-bar output projection -> concatenated x (B*N, 768)
  for (int bar = 0; bar < 3; ++bar) {
    gemm_flat_kernel<false, false><<<gp, bp, 0, stream>>>(
        oB + (size_t)bar * MTOT * E_, E_, E_, WoP, 16, bo, xB, 3 * E_,
        bar * E_);
  }

  // 6) h1 = relu(x @ Ws1 + bs1)   (K = 768)
  gemm_flat_kernel<true, false><<<gp, bp, 0, stream>>>(
      xB, 3 * E_, 3 * E_, Ws1P, 16, bs1, h1, E_, 0);

  // 7) out = h1 @ Ws2 + bs2  (f32 output)
  gemm_flat_kernel<false, true><<<gp, bp, 0, stream>>>(
      h1, E_, E_, Ws2P, 16, bs2, out, E_, 0);
}
